// TurboQuantKVCache_80676665688758
// MI455X (gfx1250) — matrix operand order repaired and checked by reading, not compiled, not run on hardware
//
#include <hip/hip_runtime.h>

typedef __attribute__((ext_vector_type(16))) _Float16 v16h;
typedef __attribute__((ext_vector_type(8)))  _Float16 v8h;
typedef __attribute__((ext_vector_type(8)))  float    v8f;
typedef __attribute__((ext_vector_type(4)))  float    v4f;

#define HEAD   128
#define RPAD   136                      // padded row stride in halves (16B aligned rows, 4-bank rotation)
#define WAVES  8
#define TROWS  16
#define TILE   (WAVES * TROWS)          // 128 rows per block tile
#define QEPS   1e-8f

#define RM_BYTES  (HEAD * RPAD * 2)     // 34816
#define SMEM_SZ   (3 * RM_BYTES)        // Rm + Rt + CQ = 104448 bytes

__global__ __launch_bounds__(256, 1) void tq_kernel(
    const float* __restrict__ kv,
    const float* __restrict__ rotg,
    const float* __restrict__ bnd,
    const float* __restrict__ cpt,
    float* __restrict__ out,
    int nrows)
{
    extern __shared__ char smem[];
    _Float16* Rm = (_Float16*)(smem);                 // Rm[r*RPAD + c] = R[r][c]  (B for GEMM2: R^T)
    _Float16* Rt = (_Float16*)(smem + RM_BYTES);      // Rt[c*RPAD + r] = R[r][c]  (B for GEMM1: R)
    _Float16* CQ = (_Float16*)(smem + 2 * RM_BYTES);  // per-wave quantized tiles

    const int tid  = threadIdx.x;
    const int wave = tid >> 5;
    const int lane = tid & 31;
    const int hi   = lane >> 4;   // half-wave select (K/M sub-block)
    const int m    = lane & 15;   // row index within A tile / column index within B tile

    // ---- stage rotation into LDS in both orders, f32 -> f16 ----
    for (int i = tid; i < HEAD * HEAD; i += blockDim.x) {
        int r = i >> 7, c = i & 127;
        _Float16 h = (_Float16)rotg[i];
        Rm[r * RPAD + c] = h;
        Rt[c * RPAD + r] = h;
    }

    // quantizer constants (uniform, tiny, L2-hot)
    const float b0 = bnd[0], b1 = bnd[1], b2 = bnd[2], b3 = bnd[3],
                b4 = bnd[4], b5 = bnd[5], b6 = bnd[6];
    const float c0 = cpt[0], c1 = cpt[1], c2 = cpt[2], c3 = cpt[3],
                c4 = cpt[4], c5 = cpt[5], c6 = cpt[6], c7 = cpt[7];

    __syncthreads();

    _Float16* cq = CQ + wave * TROWS * RPAD;
    const int ntiles = nrows >> 7;

    for (int tile = blockIdx.x; tile < ntiles; tile += gridDim.x) {
        const int row0 = tile * TILE + wave * TROWS;
        const float* __restrict__ rowp = kv + (size_t)(row0 + m) * HEAD;

        // ---- A fragments for GEMM1: f32 global -> f16 regs (16x32 A layout) ----
        v16h A[4];
        #pragma unroll
        for (int kt = 0; kt < 4; ++kt) {
            const int k0 = kt * 32 + hi * 8;
            v4f p0 = *(const v4f*)(rowp + k0);
            v4f p1 = *(const v4f*)(rowp + k0 + 4);
            v4f p2 = *(const v4f*)(rowp + k0 + 16);
            v4f p3 = *(const v4f*)(rowp + k0 + 20);
            #pragma unroll
            for (int j = 0; j < 4; ++j) {
                A[kt][j]      = (_Float16)p0[j];
                A[kt][4 + j]  = (_Float16)p1[j];
                A[kt][8 + j]  = (_Float16)p2[j];
                A[kt][12 + j] = (_Float16)p3[j];
            }
        }

        // ---- GEMM1: rotated = kv_tile * R ----
        float vals[8][8];                       // [nt][v] rotated values in D layout
        float rowSq[8] = {0,0,0,0,0,0,0,0};
        #pragma unroll
        for (int nt = 0; nt < 8; ++nt) {
            v8f acc = {0.f,0.f,0.f,0.f,0.f,0.f,0.f,0.f};
            #pragma unroll
            for (int kt = 0; kt < 4; ++kt) {
                const _Float16* bp = Rt + (nt * 16 + m) * RPAD + kt * 32 + hi * 8;
                v8h blo = *(const v8h*)bp;
                v8h bhi = *(const v8h*)(bp + 16);
                v16h B;
                #pragma unroll
                for (int j = 0; j < 8; ++j) { B[j] = blo[j]; B[8 + j] = bhi[j]; }
                acc = __builtin_amdgcn_wmma_f32_16x16x32_f16(
                          false, A[kt], false, B, (short)0, acc, false, false);
            }
            #pragma unroll
            for (int v = 0; v < 8; ++v) {
                float x = acc[v];
                vals[nt][v] = x;
                rowSq[v] += x * x;
            }
        }

        // ---- per-row radius: reduce across the 16 lanes of each half-wave ----
        float rinv[8], rq[8];
        #pragma unroll
        for (int v = 0; v < 8; ++v) {
            float s = rowSq[v];
            s += __shfl_xor(s, 1, 32);
            s += __shfl_xor(s, 2, 32);
            s += __shfl_xor(s, 4, 32);
            s += __shfl_xor(s, 8, 32);
            float r = fmaxf(sqrtf(s), QEPS);
            rinv[v] = 1.0f / r;
            rq[v]   = (float)(_Float16)r;       // radius as stored in fp16
        }

        // ---- quantize unit vector to codepoints; accumulate ||u||^2 ----
        float cSq[8] = {0,0,0,0,0,0,0,0};
        #pragma unroll
        for (int nt = 0; nt < 8; ++nt) {
            #pragma unroll
            for (int v = 0; v < 8; ++v) {
                float x  = vals[nt][v] * rinv[v];
                float cv = c0;                   // idx = count(boundaries < x)
                cv = (x > b0) ? c1 : cv;
                cv = (x > b1) ? c2 : cv;
                cv = (x > b2) ? c3 : cv;
                cv = (x > b3) ? c4 : cv;
                cv = (x > b4) ? c5 : cv;
                cv = (x > b5) ? c6 : cv;
                cv = (x > b6) ? c7 : cv;
                vals[nt][v] = cv;
                cSq[v] += cv * cv;
            }
        }
        float scl[8];
        #pragma unroll
        for (int v = 0; v < 8; ++v) {
            float s = cSq[v];
            s += __shfl_xor(s, 1, 32);
            s += __shfl_xor(s, 2, 32);
            s += __shfl_xor(s, 4, 32);
            s += __shfl_xor(s, 8, 32);
            scl[v] = rq[v] / fmaxf(sqrtf(s), QEPS);
        }

        // ---- write dequantized tile to LDS (row-major f16) ----
        #pragma unroll
        for (int nt = 0; nt < 8; ++nt) {
            #pragma unroll
            for (int v = 0; v < 8; ++v) {
                cq[(v + 8 * hi) * RPAD + nt * 16 + m] =
                    (_Float16)(vals[nt][v] * scl[v]);
            }
        }
        __syncthreads();

        // ---- A fragments for GEMM2 from LDS ----
        v16h A2[4];
        #pragma unroll
        for (int kt = 0; kt < 4; ++kt) {
            const int k0 = kt * 32 + hi * 8;
            const _Float16* ap = cq + m * RPAD + k0;
            v8h lo  = *(const v8h*)ap;
            v8h hi8 = *(const v8h*)(ap + 16);
            #pragma unroll
            for (int j = 0; j < 8; ++j) { A2[kt][j] = lo[j]; A2[kt][8 + j] = hi8[j]; }
        }

        // ---- GEMM2: out = q_tile * R^T ; store straight to global ----
        #pragma unroll
        for (int nt = 0; nt < 8; ++nt) {
            v8f acc = {0.f,0.f,0.f,0.f,0.f,0.f,0.f,0.f};
            #pragma unroll
            for (int kt = 0; kt < 4; ++kt) {
                const _Float16* bp = Rm + (nt * 16 + m) * RPAD + kt * 32 + hi * 8;
                v8h blo = *(const v8h*)bp;
                v8h bhi = *(const v8h*)(bp + 16);
                v16h B;
                #pragma unroll
                for (int j = 0; j < 8; ++j) { B[j] = blo[j]; B[8 + j] = bhi[j]; }
                acc = __builtin_amdgcn_wmma_f32_16x16x32_f16(
                          false, A2[kt], false, B, (short)0, acc, false, false);
            }
            #pragma unroll
            for (int v = 0; v < 8; ++v) {
                out[(size_t)(row0 + v + 8 * hi) * HEAD + nt * 16 + m] = acc[v];
            }
        }
        __syncthreads();
    }
}

extern "C" void kernel_launch(void* const* d_in, const int* in_sizes, int n_in,
                              void* d_out, int out_size, void* d_ws, size_t ws_size,
                              hipStream_t stream) {
    const float* kv  = (const float*)d_in[0];
    const float* rot = (const float*)d_in[1];
    const float* bnd = (const float*)d_in[2];
    const float* cpt = (const float*)d_in[3];
    float* out = (float*)d_out;

    const int nrows  = in_sizes[0] / HEAD;        // 524288 for the given shapes
    int ntiles = nrows >> 7;
    if (ntiles < 1) ntiles = 1;
    int grid = ntiles < 1024 ? ntiles : 1024;

    (void)hipFuncSetAttribute((const void*)tq_kernel,
                              hipFuncAttributeMaxDynamicSharedMemorySize, SMEM_SZ);
    tq_kernel<<<grid, 256, SMEM_SZ, stream>>>(kv, rot, bnd, cpt, out, nrows);
}
